// TriFeaPred_41532333752468
// MI455X (gfx1250) — matrix operand order repaired and checked by reading, not compiled, not run on hardware
//
#include <hip/hip_runtime.h>
#include <math.h>

#define BSZ   2
#define NP    4096
#define KTOP  33
#define WAVES 4

typedef __attribute__((ext_vector_type(16))) __bf16 v16bf;
typedef __attribute__((ext_vector_type(8)))  __bf16 v8bf;
typedef __attribute__((ext_vector_type(8)))  float  v8f;

// f32 -> bf16 raw bits, round-to-nearest-even
__device__ __forceinline__ unsigned short f2h(float f) {
  union { float f; unsigned int u; } in; in.f = f;
  unsigned int u = in.u;
  u += 0x7FFFu + ((u >> 16) & 1u);
  return (unsigned short)(u >> 16);
}
__device__ __forceinline__ float h2f(unsigned short s) {
  union { unsigned int u; float f; } o; o.u = ((unsigned int)s) << 16;
  return o.f;
}
union V16 { v16bf v; v8bf h[2]; };

// ---------------------------------------------------------------------------
// Branchless WMMA GEMM, 32x32 output tile per wave (2x2 register tiling):
//   A   [M x Kp]   bf16, zero-padded cols; M % 128 == 0
//   Wp  [Ncp x Kp] bf16, zero-padded packed weights
// Per 32-K step: 8x global_load_b128 feed 4x v_wmma (2x fragment reuse).
// ---------------------------------------------------------------------------
__global__ void __launch_bounds__(WAVES * 32)
gemm_bf16(const unsigned short* __restrict__ A, int Kp,
          const unsigned short* __restrict__ Wp,
          const float* __restrict__ pb, const float* __restrict__ ps,
          const float* __restrict__ psh,
          unsigned short* __restrict__ outb, float* __restrict__ outf,
          int ldo, int Nc, int relu) {
  const int lane = threadIdx.x & 31;
  const int wave = threadIdx.x >> 5;
  const int hf   = lane >> 4;       // lane half
  const int lq   = lane & 15;
  const int tile_m = (blockIdx.x * WAVES + wave) * 32;
  const int tile_n = blockIdx.y * 32;

  const unsigned short* arow0 = A  + (size_t)(tile_m + lq) * Kp + hf * 8;
  const unsigned short* arow1 = arow0 + (size_t)16 * Kp;
  const unsigned short* wrow0 = Wp + (size_t)(tile_n + lq) * Kp + hf * 8;
  const unsigned short* wrow1 = wrow0 + (size_t)16 * Kp;

  v8f acc00 = {0.f,0.f,0.f,0.f,0.f,0.f,0.f,0.f};
  v8f acc01 = acc00, acc10 = acc00, acc11 = acc00;

  for (int kb = 0; kb < Kp; kb += 32) {
    V16 a0, a1, b0, b1;
    a0.h[0] = *(const v8bf*)(arow0 + kb);        // K = kb + hf*8 + 0..7
    a0.h[1] = *(const v8bf*)(arow0 + kb + 16);   // K = kb + 16 + hf*8 + 0..7
    a1.h[0] = *(const v8bf*)(arow1 + kb);
    a1.h[1] = *(const v8bf*)(arow1 + kb + 16);
    b0.h[0] = *(const v8bf*)(wrow0 + kb);        // K = kb + hf*8 + 0..7
    b0.h[1] = *(const v8bf*)(wrow0 + kb + 16);   // K = kb + 16 + hf*8 + 0..7
    b1.h[0] = *(const v8bf*)(wrow1 + kb);
    b1.h[1] = *(const v8bf*)(wrow1 + kb + 16);
    acc00 = __builtin_amdgcn_wmma_f32_16x16x32_bf16(false, a0.v, false, b0.v,
                                                    (short)0, acc00, false, false);
    acc01 = __builtin_amdgcn_wmma_f32_16x16x32_bf16(false, a0.v, false, b1.v,
                                                    (short)0, acc01, false, false);
    acc10 = __builtin_amdgcn_wmma_f32_16x16x32_bf16(false, a1.v, false, b0.v,
                                                    (short)0, acc10, false, false);
    acc11 = __builtin_amdgcn_wmma_f32_16x16x32_bf16(false, a1.v, false, b1.v,
                                                    (short)0, acc11, false, false);
  }

  const int nc0 = tile_n + lq;                   // < Ncp; params zero-padded
  const int nc1 = nc0 + 16;
  const float bb0 = pb[nc0], gs0 = ps[nc0], bt0 = psh[nc0];
  const float bb1 = pb[nc1], gs1 = ps[nc1], bt1 = psh[nc1];

  if (outb) {                                    // uniform branch
#pragma unroll
    for (int r = 0; r < 8; ++r) {
      int m0 = tile_m + r + 8 * hf;              // C layout: row r + 8*half
      int m1 = m0 + 16;
      float y00 = acc00[r] + bb0, y01 = acc01[r] + bb1;
      float y10 = acc10[r] + bb0, y11 = acc11[r] + bb1;
      if (relu) {
        y00 = fmaxf(gs0 * y00 + bt0, 0.f); y01 = fmaxf(gs1 * y01 + bt1, 0.f);
        y10 = fmaxf(gs0 * y10 + bt0, 0.f); y11 = fmaxf(gs1 * y11 + bt1, 0.f);
      }
      outb[(size_t)m0 * ldo + nc0] = f2h(y00);   // pad cols naturally 0
      outb[(size_t)m0 * ldo + nc1] = f2h(y01);
      outb[(size_t)m1 * ldo + nc0] = f2h(y10);
      outb[(size_t)m1 * ldo + nc1] = f2h(y11);
    }
  } else {
#pragma unroll
    for (int r = 0; r < 8; ++r) {
      int m0 = tile_m + r + 8 * hf;
      int m1 = m0 + 16;
      float y00 = acc00[r] + bb0, y01 = acc01[r] + bb1;
      float y10 = acc10[r] + bb0, y11 = acc11[r] + bb1;
      if (relu) {
        y00 = fmaxf(gs0 * y00 + bt0, 0.f); y01 = fmaxf(gs1 * y01 + bt1, 0.f);
        y10 = fmaxf(gs0 * y10 + bt0, 0.f); y11 = fmaxf(gs1 * y11 + bt1, 0.f);
      }
      if (nc0 < Nc) {
        outf[(size_t)m0 * ldo + nc0] = y00;
        outf[(size_t)m1 * ldo + nc0] = y10;
      }
      if (nc1 < Nc) {
        outf[(size_t)m0 * ldo + nc1] = y01;
        outf[(size_t)m1 * ldo + nc1] = y11;
      }
    }
  }
}

// Pack one layer: W[Nc,K] f32 -> wp[Ncp,Kp] bf16 zero-padded; params padded.
__global__ void pack_layer(const float* __restrict__ W, const float* __restrict__ b,
                           const float* __restrict__ g, const float* __restrict__ be,
                           unsigned short* __restrict__ wp, float* __restrict__ pb,
                           float* __restrict__ ps, float* __restrict__ psh,
                           int K, int Kp, int Nc, int Ncp) {
  const size_t total = (size_t)Ncp * Kp;
  for (size_t i = (size_t)blockIdx.x * blockDim.x + threadIdx.x; i < total;
       i += (size_t)gridDim.x * blockDim.x) {
    int k = (int)(i % Kp);
    int n = (int)(i / Kp);
    wp[i] = (n < Nc && k < K) ? f2h(W[(size_t)n * K + k]) : (unsigned short)0;
    if (i < (size_t)Ncp) {
      int nn = (int)i;
      bool v = nn < Nc;
      pb[nn]  = v ? b[nn] : 0.f;
      ps[nn]  = v ? g[nn] * 0.9999950000374997f : 0.f;  // gamma / sqrt(1+1e-5)
      psh[nn] = v ? be[nn] : 0.f;
    }
  }
}

// ---------------------------------------------------------------------------
// 33-NN (self + 32 nearest ascending), shared by all SA stages.
// ---------------------------------------------------------------------------
#define KNN_TS 128
__global__ void knn_kernel(const float* __restrict__ xyz, int* __restrict__ idx_out,
                           int npts) {
  const int b = blockIdx.y;
  const int n = blockIdx.x * blockDim.x + threadIdx.x;
  const float* base = xyz + (size_t)b * npts * 3;
  const bool valid = (n < npts);
  float qx = 0.f, qy = 0.f, qz = 0.f;
  if (valid) { qx = base[n * 3 + 0]; qy = base[n * 3 + 1]; qz = base[n * 3 + 2]; }

  float bd[KTOP]; int bi[KTOP];
#pragma unroll
  for (int i = 0; i < KTOP; ++i) { bd[i] = 3.0e38f; bi[i] = 0; }

  __shared__ float tile[KNN_TS * 3];
  for (int t0 = 0; t0 < npts; t0 += KNN_TS) {
    int lim = npts - t0; if (lim > KNN_TS) lim = KNN_TS;
    for (int i = threadIdx.x; i < lim * 3; i += blockDim.x)
      tile[i] = base[(size_t)t0 * 3 + i];
    __syncthreads();
    if (valid) {
      for (int j = 0; j < lim; ++j) {
        float dx = tile[j * 3 + 0] - qx;
        float dy = tile[j * 3 + 1] - qy;
        float dz = tile[j * 3 + 2] - qz;
        float d = dx * dx + dy * dy + dz * dz;
        if (d < bd[KTOP - 1]) {
          int pos = KTOP - 1;
          while (pos > 0 && bd[pos - 1] > d) {
            bd[pos] = bd[pos - 1]; bi[pos] = bi[pos - 1]; --pos;
          }
          bd[pos] = d; bi[pos] = t0 + j;
        }
      }
    }
    __syncthreads();
  }
  if (valid) {
    size_t o = ((size_t)b * npts + n) * KTOP;
    for (int i = 0; i < KTOP; ++i) idx_out[o + i] = bi[i];
  }
}

// xyz f32 [pts,3] -> padded bf16 [pts,32]
__global__ void pad_xyz_h(const float* __restrict__ xyz, unsigned short* __restrict__ out,
                          int pts) {
  const size_t total = (size_t)pts * 32;
  for (size_t i = (size_t)blockIdx.x * blockDim.x + threadIdx.x; i < total;
       i += (size_t)gridDim.x * blockDim.x) {
    int ch = (int)(i & 31);
    size_t pt = i >> 5;
    out[i] = (ch < 3) ? f2h(xyz[pt * 3 + ch]) : (unsigned short)0;
  }
}

// out[(pt,j)] = [xyz[nbr]-xyz[pt] (3) | fea[nbr] (c)] padded to cwp, bf16
__global__ void gather_concat_h(const float* __restrict__ xyz,
                                const unsigned short* __restrict__ fea, int ldf,
                                const int* __restrict__ idx,
                                unsigned short* __restrict__ out,
                                int pts, int npts, int k, int c, int cwp) {
  const int cw = 3 + c;
  const size_t total = (size_t)pts * k * cwp;
  for (size_t i = (size_t)blockIdx.x * blockDim.x + threadIdx.x; i < total;
       i += (size_t)gridDim.x * blockDim.x) {
    int ch = (int)(i % cwp);
    size_t rc = i / cwp;
    int j  = (int)(rc % k);
    int pt = (int)(rc / k);
    unsigned short v = 0;
    if (ch < cw) {
      int bb  = pt / npts;
      int nbr = idx[(size_t)pt * KTOP + 1 + j];   // slot 0 is self
      size_t np = (size_t)bb * npts + nbr;
      if (ch < 3) v = f2h(xyz[np * 3 + ch] - xyz[(size_t)pt * 3 + ch]);
      else        v = fea[np * ldf + (ch - 3)];
    }
    out[i] = v;
  }
}

__global__ void maxpool_h(const unsigned short* __restrict__ in, int ldi,
                          unsigned short* __restrict__ out, int ldo,
                          int pts, int k, int c) {
  const size_t total = (size_t)pts * ldo;
  for (size_t i = (size_t)blockIdx.x * blockDim.x + threadIdx.x; i < total;
       i += (size_t)gridDim.x * blockDim.x) {
    int ch = (int)(i % ldo);
    size_t pt = i / ldo;
    unsigned short v = 0;
    if (ch < c) {
      float m = -3.4e38f;
      for (int j = 0; j < k; ++j)
        m = fmaxf(m, h2f(in[((size_t)pt * k + j) * ldi + ch]));
      v = f2h(m);
    }
    out[i] = v;
  }
}

__global__ void concat3_h(const unsigned short* __restrict__ p0, int c0, int ld0,
                          const unsigned short* __restrict__ p1, int c1, int ld1,
                          const unsigned short* __restrict__ p2, int c2, int ld2,
                          unsigned short* __restrict__ out, int rows, int cwp) {
  const int cw = c0 + c1 + c2;
  const size_t total = (size_t)rows * cwp;
  for (size_t i = (size_t)blockIdx.x * blockDim.x + threadIdx.x; i < total;
       i += (size_t)gridDim.x * blockDim.x) {
    int ch = (int)(i % cwp);
    size_t r = i / cwp;
    unsigned short v = 0;
    if      (ch < c0)      v = p0[r * ld0 + ch];
    else if (ch < c0 + c1) v = p1[r * ld1 + (ch - c0)];
    else if (ch < cw)      v = p2[r * ld2 + (ch - c0 - c1)];
    out[i] = v;
  }
}

__global__ void log_softmax_rows(const float* __restrict__ in, float* __restrict__ out,
                                 int rows, int c) {
  int r = blockIdx.x * blockDim.x + threadIdx.x;
  if (r >= rows) return;
  const float* x = in + (size_t)r * c;
  float m = -3.4e38f;
  for (int i = 0; i < c; ++i) m = fmaxf(m, x[i]);
  float s = 0.f;
  for (int i = 0; i < c; ++i) s += expf(x[i] - m);
  float ls = logf(s);
  for (int i = 0; i < c; ++i) out[(size_t)r * c + i] = x[i] - m - ls;
}

// ---------------------------------------------------------------------------
// Host orchestration
// ---------------------------------------------------------------------------
struct Layer {
  const float *W, *b, *g, *be;
  int cin, cout, Kp, Ncp;
  unsigned short* wp;            // packed bf16 weights [Ncp x Kp]
  float *pb, *ps, *psh;          // padded bias / scale / shift
};

static inline int idiv(int a, int b) { return (a + b - 1) / b; }
static inline int pad32(int c)       { return (c + 31) & ~31; }
static inline unsigned int gblocks(size_t total) {
  size_t g = (total + 255) / 256;
  if (g > 1048576) g = 1048576;
  if (g < 1) g = 1;
  return (unsigned int)g;
}

static void launch_gemm(const unsigned short* A, const Layer& L,
                        unsigned short* outb, float* outf, int ldo, int M,
                        int relu, hipStream_t s) {
  dim3 grid(M / (32 * WAVES), L.Ncp / 32);       // M % 128 == 0, Ncp % 32 == 0
  gemm_bf16<<<grid, dim3(WAVES * 32), 0, s>>>(A, L.Kp, L.wp, L.pb, L.ps, L.psh,
                                              outb, outf, ldo, L.cout, relu);
}

// ping-pong MLP chain; final layer optionally to bf16 `finb` or f32 `finf`.
static const unsigned short* run_mlp_h(const Layer* Ls, int nL,
                                       const unsigned short* in, int M,
                                       unsigned short* t0, unsigned short* t1,
                                       unsigned short* finb, float* finf, int ldo_f,
                                       bool final_plain, hipStream_t s) {
  const unsigned short* cur = in;
  for (int i = 0; i < nL; ++i) {
    bool last = (i == nL - 1);
    int relu  = (final_plain && last) ? 0 : 1;
    if (last && (finb || finf)) {
      launch_gemm(cur, Ls[i], finb, finf, finb ? Ls[i].Ncp : ldo_f, M, relu, s);
      cur = finb;
    } else {
      unsigned short* dst = (i & 1) ? t1 : t0;
      launch_gemm(cur, Ls[i], dst, nullptr, Ls[i].Ncp, M, relu, s);
      cur = dst;
    }
  }
  return cur;
}

extern "C" void kernel_launch(void* const* d_in, const int* in_sizes, int n_in,
                              void* d_out, int out_size, void* d_ws, size_t ws_size,
                              hipStream_t stream) {
  (void)n_in; (void)out_size; (void)ws_size;
  const float* xyz = (const float*)d_in[0];
  const int pts = BSZ * NP;                       // 8192 (multiple of 128)

  // ---- byte bump allocator on d_ws (256B aligned chunks) ----
  char* base = (char*)d_ws;
  size_t off = 0;
  auto alloc = [&](size_t bytes) -> void* {
    void* ptr = base + off;
    off += (bytes + 255) & ~(size_t)255;
    return ptr;
  };

  // ---- parse params (dict order; each layer = W,b,g,be) + pack ----
  int p = 1;
  auto grab = [&](Layer& L) {
    L.W  = (const float*)d_in[p]; int wsz = in_sizes[p]; ++p;
    L.b  = (const float*)d_in[p]; L.cout = in_sizes[p]; ++p;
    L.g  = (const float*)d_in[p]; ++p;
    L.be = (const float*)d_in[p]; ++p;
    L.cin = wsz / L.cout;
    L.Kp  = pad32(L.cin);
    L.Ncp = pad32(L.cout);
    L.wp  = (unsigned short*)alloc((size_t)L.Ncp * L.Kp * sizeof(unsigned short));
    L.pb  = (float*)alloc((size_t)L.Ncp * sizeof(float));
    L.ps  = (float*)alloc((size_t)L.Ncp * sizeof(float));
    L.psh = (float*)alloc((size_t)L.Ncp * sizeof(float));
    pack_layer<<<gblocks((size_t)L.Ncp * L.Kp), 256, 0, stream>>>(
        L.W, L.b, L.g, L.be, L.wp, L.pb, L.ps, L.psh, L.cin, L.Kp, L.cout, L.Ncp);
  };
  Layer pre[2];   for (int i = 0; i < 2; ++i) grab(pre[i]);
  Layer sa[5][3]; for (int i = 0; i < 5; ++i) for (int j = 0; j < 3; ++j) grab(sa[i][j]);
  Layer fp[5][3]; for (int i = 0; i < 5; ++i) for (int j = 0; j < 3; ++j) grab(fp[i][j]); // fp5..fp1
  Layer c12[2];   for (int i = 0; i < 2; ++i) grab(c12[i]);
  Layer eula[4];  for (int i = 0; i < 4; ++i) grab(eula[i]);
  Layer near_[4]; for (int i = 0; i < 4; ++i) grab(near_[i]);
  Layer meta[4];  for (int i = 0; i < 4; ++i) grab(meta[i]);

  // ---- activation workspace ----
  int* idxbuf = (int*)alloc((size_t)pts * KTOP * sizeof(int));
  const size_t CAP = (size_t)131072 * 256;        // worst case padded elements
  unsigned short* bufA = (unsigned short*)alloc(CAP * sizeof(unsigned short));
  unsigned short* bufB = (unsigned short*)alloc(CAP * sizeof(unsigned short));
  unsigned short* slot[7];
  for (int i = 0; i < 7; ++i)
    slot[i] = (unsigned short*)alloc((size_t)pts * 512 * sizeof(unsigned short));
  unsigned short* xyzp = (unsigned short*)alloc((size_t)pts * 32 * sizeof(unsigned short));
  float* tmpf = (float*)alloc((size_t)pts * 8 * sizeof(float));

  // ---- stage xyz as padded bf16; one shared 33-NN pass ----
  pad_xyz_h<<<gblocks((size_t)pts * 32), 256, 0, stream>>>(xyz, xyzp, pts);
  knn_kernel<<<dim3(idiv(NP, 128), BSZ), 128, 0, stream>>>(xyz, idxbuf, NP);

  // ---- pre MLP: 3 -> 8 (bn+relu) -> 16 (plain) ----
  launch_gemm(xyzp, pre[0], bufA, nullptr, pre[0].Ncp, pts, 1, stream);
  launch_gemm(bufA, pre[1], slot[0], nullptr, pre[1].Ncp, pts, 0, stream);
  int c_l[6]; c_l[0] = pre[1].cout;

  // ---- SA stages ----
  const int kk[5] = {16, 24, 32, 24, 16};
  const unsigned short* fea_in = slot[0];
  int cin = c_l[0];
  for (int si = 0; si < 5; ++si) {
    int k = kk[si];
    int M = pts * k;
    int cwp = sa[si][0].Kp;                       // pad32(3+cin)
    gather_concat_h<<<gblocks((size_t)M * cwp), 256, 0, stream>>>(
        xyz, fea_in, pad32(cin), idxbuf, bufA, pts, NP, k, cin, cwp);
    const unsigned short* mo =
        run_mlp_h(sa[si], 3, bufA, M, bufB, bufA, nullptr, nullptr, 0, false, stream);
    int cout = sa[si][2].cout;
    maxpool_h<<<gblocks((size_t)pts * pad32(cout)), 256, 0, stream>>>(
        mo, sa[si][2].Ncp, slot[1 + si], pad32(cout), pts, k, cout);
    fea_in = slot[1 + si];
    cin = cout;
    c_l[1 + si] = cout;
  }

  // ---- FP stages (fp[0]=fp5 ... fp[4]=fp1): new l_i = mlp(cat(l_i, l_{i+1})) ----
  for (int st = 0; st < 4; ++st) {
    int li = 4 - st;                               // 4,3,2,1
    int ca = c_l[li], cb = c_l[li + 1];
    int cwp = fp[st][0].Kp;                        // pad32(ca+cb)
    concat3_h<<<gblocks((size_t)pts * cwp), 256, 0, stream>>>(
        slot[li], ca, pad32(ca), slot[li + 1], cb, pad32(cb),
        nullptr, 0, 0, bufA, pts, cwp);
    run_mlp_h(fp[st], 3, bufA, pts, bufB, bufA, slot[li], nullptr, 0, false, stream);
    c_l[li] = fp[st][2].cout;
  }
  // fp1: new l0 = mlp(cat(xyz, l0, l1))
  {
    int cwp = fp[4][0].Kp;                         // pad32(3+16+448)
    concat3_h<<<gblocks((size_t)pts * cwp), 256, 0, stream>>>(
        xyzp, 3, 32, slot[0], c_l[0], pad32(c_l[0]),
        slot[1], c_l[1], pad32(c_l[1]), bufA, pts, cwp);
    run_mlp_h(fp[4], 3, bufA, pts, bufB, bufA, slot[0], nullptr, 0, false, stream);
    c_l[0] = fp[4][2].cout;
  }

  // ---- conv12: 512 -> 384 -> 256 (all bn+relu) ----
  run_mlp_h(c12, 2, slot[0], pts, bufA, bufB, slot[6], nullptr, 0, false, stream);
  const unsigned short* fea = slot[6];

  // ---- heads ----
  float* dout = (float*)d_out;
  // eula: plain final straight into d_out[0 : pts*3] (f32, unpadded)
  run_mlp_h(eula, 4, fea, pts, bufA, bufB, nullptr, dout, eula[3].cout, true, stream);
  int ce = eula[3].cout;
  // near: plain final -> f32 tmp -> log_softmax
  int cn = near_[3].cout;
  run_mlp_h(near_, 4, fea, pts, bufA, bufB, nullptr, tmpf, cn, true, stream);
  log_softmax_rows<<<idiv(pts, 256), 256, 0, stream>>>(tmpf, dout + (size_t)pts * ce,
                                                       pts, cn);
  // meta: plain final -> f32 tmp -> log_softmax
  int cm = meta[3].cout;
  run_mlp_h(meta, 4, fea, pts, bufA, bufB, nullptr, tmpf, cm, true, stream);
  log_softmax_rows<<<idiv(pts, 256), 256, 0, stream>>>(
      tmpf, dout + (size_t)pts * (ce + cn), pts, cm);
}
